// AttnReadout_2954937499918
// MI455X (gfx1250) — matrix operand order repaired and checked by reading, not compiled, not run on hardware
//
#include <hip/hip_runtime.h>
#include <hip/hip_bf16.h>

typedef __attribute__((ext_vector_type(16))) __bf16 v16bf;
typedef __attribute__((ext_vector_type(8)))  __bf16 v8bf;
typedef __attribute__((ext_vector_type(4)))  __bf16 v4bf;
typedef __attribute__((ext_vector_type(8)))  float  v8f;

#define NNODES  131072
#define DD      512
#define NGRAPH  512
#define MT      64          // rows of x per block in the GEMM
#define LDSROW  520         // bf16 elements per LDS row (512 + 8 pad -> bank-conflict-free)

// ---------- order-preserving float <-> uint encoding for atomicMax ----------
__device__ __forceinline__ unsigned enc_f32(float f) {
  unsigned u = __float_as_uint(f);
  return (u & 0x80000000u) ? ~u : (u | 0x80000000u);
}
__device__ __forceinline__ float dec_f32(unsigned u) {
  u = (u & 0x80000000u) ? (u ^ 0x80000000u) : ~u;
  return __uint_as_float(u);
}

// ---------------------------------------------------------------------------
// K0: init segment scratch (segmax encoded -inf == 0, sums/counts zero)
// ---------------------------------------------------------------------------
__global__ void init_seg_kernel(unsigned* __restrict__ segmax,
                                float* __restrict__ segsum,
                                unsigned* __restrict__ segcnt) {
  int i = blockIdx.x * blockDim.x + threadIdx.x;
  if (i < NGRAPH) { segmax[i] = 0u; segsum[i] = 0.0f; segcnt[i] = 0u; }
}

// ---------------------------------------------------------------------------
// K1: convert W (f32 row-major [n][k]) -> bf16 row-major (B-operand friendly)
// ---------------------------------------------------------------------------
__global__ void convert_w_kernel(const float* __restrict__ W,
                                 __bf16* __restrict__ Wb) {
  int i = blockIdx.x * blockDim.x + threadIdx.x;   // 0 .. 512*512-1
  if (i < DD * DD) Wb[i] = (__bf16)W[i];
}

// ---------------------------------------------------------------------------
// K2: fused GEMM + tanh + dot(query): score[i] = sum_n tanh((x Wt)[i,n]+b[n])*q[n]
// Block: 128 threads = 4 waves; wave w owns M rows [blk*64 + w*16, +16).
// A (x tile, bf16) staged in LDS then register-cached for the whole K=512;
// B (W rows as K-major columns) streamed from L2 as bf16.
// ---------------------------------------------------------------------------
__global__ __launch_bounds__(128)
void gemm_score_kernel(const float* __restrict__ x,
                       const __bf16* __restrict__ Wb,
                       const float* __restrict__ bias,
                       const float* __restrict__ query,
                       float* __restrict__ score) {
  extern __shared__ __bf16 lds[];     // MT * LDSROW bf16 = 66,560 bytes

  const int tid  = threadIdx.x;       // 0..127
  const int lane = tid & 31;
  const int wave = tid >> 5;          // 0..3
  const int hi   = lane >> 4;         // lane half (K-half selector)
  const long rowBase = (long)blockIdx.x * MT;

  // ---- stage 64x512 f32 tile -> bf16 LDS (float4 loads, 8B ds stores) ----
  for (int i = tid; i < (MT * DD / 4); i += 128) {
    int row = (i * 4) >> 9;
    int col = (i * 4) & (DD - 1);
    float4 v = ((const float4*)x)[(rowBase + row) * (DD / 4) + (col >> 2)];
    v4bf h;
    h[0] = (__bf16)v.x; h[1] = (__bf16)v.y; h[2] = (__bf16)v.z; h[3] = (__bf16)v.w;
    *(v4bf*)&lds[row * LDSROW + col] = h;
  }
  __syncthreads();

  // ---- register-cache all 16 A tiles (16x32 bf16 each) for this wave ----
  // A layout (matches B): lanes0-15 M=lane hold K=[k0..k0+15];
  // lanes16-31 hold K=[k0+16..k0+31].
  const int mloc = wave * 16 + (lane & 15);
  v16bf a_t[16];
#pragma unroll
  for (int t = 0; t < 16; ++t) {
    const int k0 = t * 32;
    v8bf alo = *(const v8bf*)&lds[mloc * LDSROW + k0 + hi * 16];
    v8bf ahi = *(const v8bf*)&lds[mloc * LDSROW + k0 + hi * 16 + 8];
    a_t[t] = __builtin_shufflevector(alo, ahi,
                                     0, 1, 2, 3, 4, 5, 6, 7,
                                     8, 9, 10, 11, 12, 13, 14, 15);
  }

  // per-lane partial scores for the 8 C rows this lane owns (M = r + 8*hi)
  float sc[8];
#pragma unroll
  for (int r = 0; r < 8; ++r) sc[r] = 0.0f;

  // ---- sweep all 32 N-tiles; full-K accumulate then fuse tanh * q ----
  for (int n = 0; n < 32; ++n) {
    const int ncol = n * 16 + (lane & 15);
    // B 32x16 layout: lane holds 16 consecutive K of column ncol,
    // K-offset hi*16; contiguous 32B in bf16 W row.
    const __bf16* wrow = Wb + (long)ncol * DD + hi * 16;

    v8f c = {};
#pragma unroll
    for (int t = 0; t < 16; ++t) {
      v16bf bt = *(const v16bf*)(wrow + t * 32);   // 32B, 32B-aligned
      c = __builtin_amdgcn_wmma_f32_16x16x32_bf16(
              /*neg_a=*/false, a_t[t],
              /*neg_b=*/false, bt,
              /*c_mod=*/(short)0, c,
              /*reuse_a=*/false, /*reuse_b=*/false);
    }

    const float bv = bias[ncol];
    const float qv = query[ncol];
#pragma unroll
    for (int r = 0; r < 8; ++r)
      sc[r] += tanhf(c[r] + bv) * qv;
  }

  // ---- reduce partial scores across the 16 lanes sharing each M row ----
#pragma unroll
  for (int r = 0; r < 8; ++r) {
    sc[r] += __shfl_xor(sc[r], 1, 32);
    sc[r] += __shfl_xor(sc[r], 2, 32);
    sc[r] += __shfl_xor(sc[r], 4, 32);
    sc[r] += __shfl_xor(sc[r], 8, 32);
  }
  if ((lane & 15) == 0) {
#pragma unroll
    for (int r = 0; r < 8; ++r)
      score[rowBase + wave * 16 + hi * 8 + r] = sc[r];
  }
}

// ---------------------------------------------------------------------------
// K3: per-node atomic max of score into segment + segment counts
// ---------------------------------------------------------------------------
__global__ void seg_max_cnt_kernel(const float* __restrict__ score,
                                   const long long* __restrict__ gp,
                                   unsigned* __restrict__ segmax,
                                   unsigned* __restrict__ segcnt) {
  int i = blockIdx.x * blockDim.x + threadIdx.x;
  if (i < NNODES) {
    int s = (int)gp[i];
    atomicMax(&segmax[s], enc_f32(score[i]));
    atomicAdd(&segcnt[s], 1u);
  }
}

// ---------------------------------------------------------------------------
// K4: exclusive prefix scan of 512 counts -> segment start offsets (trivial)
// ---------------------------------------------------------------------------
__global__ void scan_kernel(const unsigned* __restrict__ segcnt,
                            unsigned* __restrict__ segstart) {
  if (threadIdx.x == 0 && blockIdx.x == 0) {
    unsigned acc = 0;
    for (int s = 0; s < NGRAPH; ++s) { segstart[s] = acc; acc += segcnt[s]; }
    segstart[NGRAPH] = acc;
  }
}

// ---------------------------------------------------------------------------
// K5: e = exp(score - segmax[seg]); segment-sum the e's
// ---------------------------------------------------------------------------
__global__ void exp_sum_kernel(const float* __restrict__ score,
                               const long long* __restrict__ gp,
                               const unsigned* __restrict__ segmax,
                               float* __restrict__ evec,
                               float* __restrict__ segsum) {
  int i = blockIdx.x * blockDim.x + threadIdx.x;
  if (i < NNODES) {
    int s = (int)gp[i];
    float e = __expf(score[i] - dec_f32(segmax[s]));
    evec[i] = e;
    atomicAdd(&segsum[s], e);
  }
}

// ---------------------------------------------------------------------------
// K6: readout: out[g, :] = sum_{i in seg g} (e_i / denom_g) * x[i, :]
// one block per graph; thread t owns dims 2t, 2t+1 (coalesced float2 loads)
// ---------------------------------------------------------------------------
__global__ __launch_bounds__(256)
void readout_kernel(const float* __restrict__ x,
                    const float* __restrict__ evec,
                    const unsigned* __restrict__ segstart,
                    const unsigned* __restrict__ segcnt,
                    const float* __restrict__ segsum,
                    float* __restrict__ out) {
  const int g = blockIdx.x;
  const unsigned s0  = segstart[g];
  const unsigned cnt = segcnt[g];
  const float denom  = segsum[g];
  const float inv    = (denom > 0.0f) ? (1.0f / denom) : 0.0f;
  const int d0 = threadIdx.x * 2;

  float2 acc = make_float2(0.0f, 0.0f);
  for (unsigned i = 0; i < cnt; ++i) {
    const long node = (long)(s0 + i);
    const float w = evec[node] * inv;
    float2 xv = *(const float2*)&x[node * DD + d0];
    acc.x += w * xv.x;
    acc.y += w * xv.y;
  }
  *(float2*)&out[(long)g * DD + d0] = acc;
}

// ---------------------------------------------------------------------------
extern "C" void kernel_launch(void* const* d_in, const int* in_sizes, int n_in,
                              void* d_out, int out_size, void* d_ws, size_t ws_size,
                              hipStream_t stream) {
  const float*     x  = (const float*)d_in[0];
  const long long* gp = (const long long*)d_in[1];   // int64 graph_ptr
  const float*     W  = (const float*)d_in[2];
  const float*     b  = (const float*)d_in[3];
  const float*     q  = (const float*)d_in[4];
  float* out = (float*)d_out;

  // workspace layout (bytes):
  char* ws = (char*)d_ws;
  float*    score    = (float*)(ws + 0);                 // 131072 f32
  float*    evec     = (float*)(ws + 524288);            // 131072 f32
  __bf16*   Wb       = (__bf16*)(ws + 1048576);          // 512*512 bf16
  unsigned* segmax   = (unsigned*)(ws + 1572864);        // 512
  float*    segsum   = (float*)  (ws + 1572864 + 2048);  // 512
  unsigned* segcnt   = (unsigned*)(ws + 1572864 + 4096); // 512
  unsigned* segstart = (unsigned*)(ws + 1572864 + 6144); // 513

  // K0: init segment scratch
  init_seg_kernel<<<(NGRAPH + 255) / 256, 256, 0, stream>>>(segmax, segsum, segcnt);

  // K1: W -> bf16
  convert_w_kernel<<<(DD * DD + 255) / 256, 256, 0, stream>>>(W, Wb);

  // K2: fused GEMM(tanh)+dot -> score, bf16 WMMA
  const size_t lds_bytes = (size_t)MT * LDSROW * sizeof(__bf16); // 66,560 B
  gemm_score_kernel<<<NNODES / MT, 128, lds_bytes, stream>>>(x, Wb, b, q, score);

  // K3: segment max + counts
  seg_max_cnt_kernel<<<(NNODES + 255) / 256, 256, 0, stream>>>(score, gp, segmax, segcnt);

  // K4: prefix scan -> segment starts
  scan_kernel<<<1, 32, 0, stream>>>(segcnt, segstart);

  // K5: exp + segment sum
  exp_sum_kernel<<<(NNODES + 255) / 256, 256, 0, stream>>>(score, gp, segmax, evec, segsum);

  // K6: weighted readout (writes every out element; no init needed)
  readout_kernel<<<NGRAPH, 256, 0, stream>>>(x, evec, segstart, segcnt, segsum, out);
}
